// BeliefStateTracker_9002251452626
// MI455X (gfx1250) — matrix operand order repaired and checked by reading, not compiled, not run on hardware
//
#include <hip/hip_runtime.h>
#include <hip/hip_bf16.h>
#include <math.h>
#include <stdint.h>

// ---------------------------------------------------------------------------
// BeliefStateTracker fused pipeline for gfx1250 (CDNA5 / MI455X)
// B=32, N=1024, H=256, A=32, O=256.  All GEMMs via v_wmma_f32_16x16x32_f16.
// Round 3: same 64-row blocking / B-fragment reuse as round 2, but register
// pressure reduced under 256 VGPRs: stage-1 gate GEMMs serialized and the
// update gate parked in the (still dead) nxt region of the z buffer in LDS
// instead of 64 live VGPRs.  Removes s_set_vgpr_msb churn + SGPR spills.
// ---------------------------------------------------------------------------

typedef _Float16 v16h __attribute__((ext_vector_type(16)));
typedef _Float16 v8h  __attribute__((ext_vector_type(8)));
typedef float    v8f  __attribute__((ext_vector_type(8)));

#define BB    32
#define NN    1024
#define HH    256
#define AA    32
#define OO    256
#define KHA   288   // H + A
#define KHO   512   // H + O
#define SX    304   // LDS row stride (halves) for x / xr / h buffers (288 + pad)
#define SZ    528   // LDS row stride for z buffer (512 + pad)
#define MROWS 64    // rows (particles) per block
#define RT    4     // row tiles of 16 per block
#define EPSF  1e-10f

__device__ __forceinline__ float sigmoidf_(float x) { return 1.0f / (1.0f + expf(-x)); }
__device__ __forceinline__ float gelu_exact_(float x) {
  return 0.5f * x * (1.0f + erff(x * 0.70710678118654752f));
}

// 4 row-tiles x 1 col-tile of  D = A(64xK, f16 LDS) * W^T(col-major f16).
// B fragment loaded once per k-step and reused for all 4 row tiles.
// A fragment layout used here (16-bit A 16x32):
//   lane(0..15)=row, K = kk+laneHi*16..+15 (contiguous, matches B)
// B fragment (col-major weights => contiguous per lane):
//   lanes0-15: col=colBase+lane, K=kk..kk+15 ; lanes16-31: col=colBase+lane-16,
//   K=kk+16..kk+31
__device__ __forceinline__ void wmma_block4(const _Float16* __restrict__ Abuf,
                                            int strideA, int K,
                                            const _Float16* __restrict__ WT,
                                            int colBase, int lane, v8f acc[RT]) {
  const int l15 = lane & 15;
  const int lhi = lane >> 4;
#pragma unroll
  for (int rt = 0; rt < RT; ++rt) acc[rt] = (v8f){};
  const _Float16* __restrict__ bcol = WT + (size_t)(colBase + l15) * K + lhi * 16;
  const _Float16* __restrict__ abase = Abuf + l15 * strideA + lhi * 16;
  for (int kk = 0; kk < K; kk += 32) {
    v16h b = *(const v16h*)(bcol + kk);
#pragma unroll
    for (int rt = 0; rt < RT; ++rt) {
      const _Float16* arow = abase + (rt * 16) * strideA;
      v8h alo = *(const v8h*)(arow + kk);
      v8h ahi = *(const v8h*)(arow + kk + 8);
      v16h a;
#pragma unroll
      for (int i = 0; i < 8; ++i) { a[i] = alo[i]; a[i + 8] = ahi[i]; }
      acc[rt] = __builtin_amdgcn_wmma_f32_16x16x32_f16(false, a, false, b, (short)0,
                                                       acc[rt], false, false);
    }
  }
}

// ---------------------------------------------------------------------------
// Prep: W[k][n] (f32, KxN row-major) -> WT[n][k] (f16, col-major by n)
// ---------------------------------------------------------------------------
__global__ void bst_transpose_f32_f16(const float* __restrict__ src,
                                      _Float16* __restrict__ dst, int K, int Nc) {
  int i = blockIdx.x * blockDim.x + threadIdx.x;
  if (i < K * Nc) {
    int k = i / Nc;
    int n = i - k * Nc;
    dst[(size_t)n * K + k] = (_Float16)src[i];
  }
}

// ---------------------------------------------------------------------------
// Phase A: fused GRU transition + observation MLP.
// One block = 64 particle rows. 8 waves x (2 col tiles x 4 row tiles).
// LDS aliasing: bufA = x then h1 ; bufB = x_reset then h0 ;
//               sh_z cols 0..255 = update-gate then nxt.
// ---------------------------------------------------------------------------
__global__ __launch_bounds__(256) void bst_gemm_kernel(
    const float* __restrict__ particles, const float* __restrict__ action,
    const float* __restrict__ observation,
    const _Float16* __restrict__ WrT, const _Float16* __restrict__ WuT,
    const _Float16* __restrict__ WcT, const _Float16* __restrict__ W0T,
    const _Float16* __restrict__ W1T,
    const float* __restrict__ br, const float* __restrict__ bu,
    const float* __restrict__ bc, const float* __restrict__ b0,
    const float* __restrict__ b1, const float* __restrict__ W2,
    const float* __restrict__ b2,
    float* __restrict__ ws_nxt, float* __restrict__ ws_loglik) {
  __shared__ __align__(16) _Float16 bufA[MROWS * SX];  // x -> h1
  __shared__ __align__(16) _Float16 bufB[MROWS * SX];  // x_reset -> h0
  __shared__ __align__(16) _Float16 sh_z[MROWS * SZ];  // [ugate->nxt | obs]
  __shared__ float sh_red[256];

  _Float16* sh_x  = bufA;
  _Float16* sh_h1 = bufA;
  _Float16* sh_xr = bufB;
  _Float16* sh_h0 = bufB;

  const int tid = threadIdx.x;
  const int blk = blockIdx.x;
  const int b = blk >> 4;                 // NN/MROWS = 16 row-groups per batch
  const int g = blk & 15;
  const int rowBase = b * NN + g * MROWS; // flattened [B*N] row index

  // ---- stage 0: stage inputs into LDS as f16 ----
  for (int i = tid; i < MROWS * HH; i += 256) {
    int r = i >> 8, c = i & 255;
    sh_x[r * SX + c] = (_Float16)particles[(size_t)(rowBase + r) * HH + c];
  }
  for (int i = tid; i < MROWS * AA; i += 256) {
    int r = i >> 5, c = i & 31;
    _Float16 v = (_Float16)action[b * AA + c];
    sh_x[r * SX + HH + c] = v;
    sh_xr[r * SX + HH + c] = v;
  }
  for (int i = tid; i < MROWS * OO; i += 256) {
    int r = i >> 8, c = i & 255;
    sh_z[r * SZ + HH + c] = (_Float16)observation[b * OO + c];
  }
  __syncthreads();

  const int wave = tid >> 5;
  const int lane = tid & 31;
  const int l15 = lane & 15;
  const int lhi = lane >> 4;

  // ---- stage 1: gates.  Pass 1: update gate -> sh_z (f16, cols 0..255).
  //                        Pass 2: reset gate -> x_reset in bufB.
  // Only one 4-tile accumulator array live at a time (register pressure).
#pragma unroll
  for (int tt = 0; tt < 2; ++tt) {
    const int colBase = (wave * 2 + tt) * 16;
    const int col = colBase + l15;
    {
      v8f aU[RT];
      wmma_block4(sh_x, SX, KHA, WuT, colBase, lane, aU);
      const float buv = bu[col];
#pragma unroll
      for (int rt = 0; rt < RT; ++rt) {
#pragma unroll
        for (int i = 0; i < 8; ++i) {
          const int row = rt * 16 + i + lhi * 8;
          sh_z[row * SZ + col] = (_Float16)sigmoidf_(aU[rt][i] + buv);
        }
      }
    }
    {
      v8f aR[RT];
      wmma_block4(sh_x, SX, KHA, WrT, colBase, lane, aR);
      const float brv = br[col];
#pragma unroll
      for (int rt = 0; rt < RT; ++rt) {
#pragma unroll
        for (int i = 0; i < 8; ++i) {
          const int row = rt * 16 + i + lhi * 8;
          float rr = sigmoidf_(aR[rt][i] + brv);
          sh_xr[row * SX + col] = (_Float16)((float)sh_x[row * SX + col] * rr);
        }
      }
    }
  }
  __syncthreads();

  // ---- stage 2: candidate gemm, nxt = (1-u)p + u*cand.
  // Reads the parked u-gate from sh_z[row][col], then overwrites the same
  // slot with nxt (same lane wrote it in stage 1; ordered within thread).
#pragma unroll
  for (int tt = 0; tt < 2; ++tt) {
    const int colBase = (wave * 2 + tt) * 16;
    const int col = colBase + l15;
    v8f aC[RT];
    wmma_block4(sh_xr, SX, KHA, WcT, colBase, lane, aC);
    const float bcv = bc[col];
#pragma unroll
    for (int rt = 0; rt < RT; ++rt) {
#pragma unroll
      for (int i = 0; i < 8; ++i) {
        const int row = rt * 16 + i + lhi * 8;
        float cand = tanhf(aC[rt][i] + bcv);
        float p = (float)sh_x[row * SX + col];
        float uu = (float)sh_z[row * SZ + col];
        float nx = (1.0f - uu) * p + uu * cand;
        ws_nxt[(size_t)(rowBase + row) * HH + col] = nx;
        sh_z[row * SZ + col] = (_Float16)nx;
      }
    }
  }
  __syncthreads();

  // ---- stage 3: h0 = gelu(z @ W0 + b0), K = 512 (writes bufB) ----
#pragma unroll
  for (int tt = 0; tt < 2; ++tt) {
    const int colBase = (wave * 2 + tt) * 16;
    const int col = colBase + l15;
    v8f a0[RT];
    wmma_block4(sh_z, SZ, KHO, W0T, colBase, lane, a0);
    const float b0v = b0[col];
#pragma unroll
    for (int rt = 0; rt < RT; ++rt) {
#pragma unroll
      for (int i = 0; i < 8; ++i) {
        const int row = rt * 16 + i + lhi * 8;
        sh_h0[row * SX + col] = (_Float16)gelu_exact_(a0[rt][i] + b0v);
      }
    }
  }
  __syncthreads();

  // ---- stage 4: h1 = gelu(h0 @ W1 + b1), K = 256 (writes bufA) ----
#pragma unroll
  for (int tt = 0; tt < 2; ++tt) {
    const int colBase = (wave * 2 + tt) * 16;
    const int col = colBase + l15;
    v8f a1[RT];
    wmma_block4(sh_h0, SX, HH, W1T, colBase, lane, a1);
    const float b1v = b1[col];
#pragma unroll
    for (int rt = 0; rt < RT; ++rt) {
#pragma unroll
      for (int i = 0; i < 8; ++i) {
        const int row = rt * 16 + i + lhi * 8;
        sh_h1[row * SX + col] = (_Float16)gelu_exact_(a1[rt][i] + b1v);
      }
    }
  }
  __syncthreads();

  // ---- stage 5: log_lik = h1 @ W2 + b2 (N=1 -> VALU dot + LDS reduce) ----
  {
    const int r = tid >> 2;   // row 0..63
    const int kc = tid & 3;   // k-chunk 0..3 (64 k each)
    float part = 0.0f;
#pragma unroll
    for (int j = 0; j < 64; ++j)
      part += (float)sh_h1[r * SX + kc * 64 + j] * W2[kc * 64 + j];
    sh_red[tid] = part;
  }
  __syncthreads();
  if (tid < MROWS) {
    float s = b2[0];
#pragma unroll
    for (int j = 0; j < 4; ++j) s += sh_red[tid * 4 + j];
    ws_loglik[rowBase + tid] = s;
  }
}

// ---------------------------------------------------------------------------
// Phase B: per-batch stable softmax, ESS, resample flag, inclusive cumsum.
// ---------------------------------------------------------------------------
__global__ __launch_bounds__(1024) void bst_weights_kernel(
    const float* __restrict__ weights, const float* __restrict__ ws_loglik,
    float* __restrict__ ws_neww, float* __restrict__ ws_csum,
    int* __restrict__ ws_should, float* __restrict__ out_ess) {
  __shared__ float sd[NN];
  const int b = blockIdx.x;
  const int tid = threadIdx.x;

  float lw = logf(weights[b * NN + tid] + EPSF) + ws_loglik[b * NN + tid];

  sd[tid] = lw; __syncthreads();
  for (int s = NN / 2; s > 0; s >>= 1) {
    if (tid < s) sd[tid] = fmaxf(sd[tid], sd[tid + s]);
    __syncthreads();
  }
  const float mx = sd[0]; __syncthreads();

  const float ex = expf(lw - mx);
  sd[tid] = ex; __syncthreads();
  for (int s = NN / 2; s > 0; s >>= 1) {
    if (tid < s) sd[tid] += sd[tid + s];
    __syncthreads();
  }
  const float sum = sd[0]; __syncthreads();
  const float nw = ex / sum;
  ws_neww[b * NN + tid] = nw;

  sd[tid] = nw * nw; __syncthreads();
  for (int s = NN / 2; s > 0; s >>= 1) {
    if (tid < s) sd[tid] += sd[tid + s];
    __syncthreads();
  }
  const float ess = 1.0f / (sd[0] + EPSF);
  __syncthreads();
  if (tid == 0) {
    out_ess[b] = ess;
    ws_should[b] = (ess < 0.5f * (float)NN) ? 1 : 0;
  }

  sd[tid] = nw; __syncthreads();
  for (int off = 1; off < NN; off <<= 1) {
    float v = sd[tid];
    if (tid >= off) v += sd[tid - off];
    __syncthreads();
    sd[tid] = v;
    __syncthreads();
  }
  ws_csum[b * NN + tid] = sd[tid];
}

// ---------------------------------------------------------------------------
// Phase C: systematic resampling + final weights + belief reduction.
// ---------------------------------------------------------------------------
__global__ __launch_bounds__(256) void bst_resample_kernel(
    const float* __restrict__ ws_nxt, const float* __restrict__ ws_neww,
    const float* __restrict__ ws_csum, const int* __restrict__ ws_should,
    const float* __restrict__ u,
    float* __restrict__ out_belief, float* __restrict__ out_nxt,
    float* __restrict__ out_neww) {
  __shared__ int sh_idx[NN];
  __shared__ float sh_w[NN];
  const int b = blockIdx.x;
  const int tid = threadIdx.x;
  const int should = ws_should[b];
  const float uu = u[b];
  const float* __restrict__ cs = ws_csum + (size_t)b * NN;

  for (int j = tid; j < NN; j += 256) {
    int src;
    float wf;
    if (should) {
      float pos = fminf(uu * (1.0f / NN) + (float)j * (1.0f / NN), 0.9999f);
      int lo = 0, hi = NN;  // searchsorted, side='left'
      while (lo < hi) {
        int mid = (lo + hi) >> 1;
        if (cs[mid] < pos) lo = mid + 1; else hi = mid;
      }
      src = (lo < NN) ? lo : (NN - 1);
      wf = 1.0f / (float)NN;
    } else {
      src = j;
      wf = ws_neww[b * NN + j];
    }
    sh_idx[j] = src;
    sh_w[j] = wf;
    out_neww[b * NN + j] = wf;
  }
  __syncthreads();

  const int c = tid;  // H column
  float bel = 0.0f;
  const float* __restrict__ src_base = ws_nxt + (size_t)b * NN * HH;
  float* __restrict__ dst_base = out_nxt + (size_t)b * NN * HH;
  for (int j = 0; j < NN; ++j) {
    float v = src_base[(size_t)sh_idx[j] * HH + c];
    dst_base[(size_t)j * HH + c] = v;
    bel += v * sh_w[j];
  }
  out_belief[b * HH + c] = bel;
}

// ---------------------------------------------------------------------------
// Launch
// ---------------------------------------------------------------------------
extern "C" void kernel_launch(void* const* d_in, const int* in_sizes, int n_in,
                              void* d_out, int out_size, void* d_ws, size_t ws_size,
                              hipStream_t stream) {
  const float* particles   = (const float*)d_in[0];
  const float* weights     = (const float*)d_in[1];
  const float* action      = (const float*)d_in[2];
  const float* observation = (const float*)d_in[3];
  const float* u           = (const float*)d_in[4];
  const float* Wr = (const float*)d_in[5];
  const float* br = (const float*)d_in[6];
  const float* Wu = (const float*)d_in[7];
  const float* bu = (const float*)d_in[8];
  const float* Wc = (const float*)d_in[9];
  const float* bc = (const float*)d_in[10];
  const float* W0 = (const float*)d_in[11];
  const float* b0 = (const float*)d_in[12];
  const float* W1 = (const float*)d_in[13];
  const float* b1 = (const float*)d_in[14];
  const float* W2 = (const float*)d_in[15];
  const float* b2 = (const float*)d_in[16];

  // ---- workspace layout ----
  char* wp = (char*)d_ws;
  _Float16* WrT = (_Float16*)wp; wp += (size_t)KHA * HH * sizeof(_Float16);
  _Float16* WuT = (_Float16*)wp; wp += (size_t)KHA * HH * sizeof(_Float16);
  _Float16* WcT = (_Float16*)wp; wp += (size_t)KHA * HH * sizeof(_Float16);
  _Float16* W0T = (_Float16*)wp; wp += (size_t)KHO * HH * sizeof(_Float16);
  _Float16* W1T = (_Float16*)wp; wp += (size_t)HH * HH * sizeof(_Float16);
  uintptr_t pa = ((uintptr_t)wp + 255u) & ~(uintptr_t)255u;
  wp = (char*)pa;
  float* ws_nxt    = (float*)wp; wp += (size_t)BB * NN * HH * sizeof(float);
  float* ws_loglik = (float*)wp; wp += (size_t)BB * NN * sizeof(float);
  float* ws_neww   = (float*)wp; wp += (size_t)BB * NN * sizeof(float);
  float* ws_csum   = (float*)wp; wp += (size_t)BB * NN * sizeof(float);
  int*   ws_should = (int*)wp;   wp += (size_t)BB * sizeof(int);

  // ---- output layout (belief, nxt, new_w, ess — flat, in return order) ----
  float* out        = (float*)d_out;
  float* out_belief = out;
  float* out_nxt    = out + (size_t)BB * HH;
  float* out_neww   = out_nxt + (size_t)BB * NN * HH;
  float* out_ess    = out_neww + (size_t)BB * NN;

  // ---- prep: f32 -> col-major f16 weights ----
  {
    int t;
    t = KHA * HH; bst_transpose_f32_f16<<<(t + 255) / 256, 256, 0, stream>>>(Wr, WrT, KHA, HH);
    t = KHA * HH; bst_transpose_f32_f16<<<(t + 255) / 256, 256, 0, stream>>>(Wu, WuT, KHA, HH);
    t = KHA * HH; bst_transpose_f32_f16<<<(t + 255) / 256, 256, 0, stream>>>(Wc, WcT, KHA, HH);
    t = KHO * HH; bst_transpose_f32_f16<<<(t + 255) / 256, 256, 0, stream>>>(W0, W0T, KHO, HH);
    t = HH * HH;  bst_transpose_f32_f16<<<(t + 255) / 256, 256, 0, stream>>>(W1, W1T, HH, HH);
  }

  // ---- phase A: fused WMMA GEMM chain (64 rows per block) ----
  bst_gemm_kernel<<<BB * NN / MROWS, 256, 0, stream>>>(
      particles, action, observation, WrT, WuT, WcT, W0T, W1T,
      br, bu, bc, b0, b1, W2, b2, ws_nxt, ws_loglik);

  // ---- phase B: softmax / ESS / cumsum ----
  bst_weights_kernel<<<BB, NN, 0, stream>>>(weights, ws_loglik, ws_neww, ws_csum,
                                            ws_should, out_ess);

  // ---- phase C: resample + belief ----
  bst_resample_kernel<<<BB, 256, 0, stream>>>(ws_nxt, ws_neww, ws_csum, ws_should, u,
                                              out_belief, out_nxt, out_neww);
}
